// HeteroTopologyEncoder_68642167324677
// MI455X (gfx1250) — matrix operand order repaired and checked by reading, not compiled, not run on hardware
//
#include <hip/hip_runtime.h>
#include <hip/hip_bf16.h>

typedef __attribute__((ext_vector_type(16))) _Float16 v16h;
typedef __attribute__((ext_vector_type(8)))  float    v8f;

#define CDIV(a, b) (((a) + (b) - 1) / (b))

// ---------------------------------------------------------------------------
// Utility: fill buffer with a constant
// ---------------------------------------------------------------------------
__global__ void fill_kernel(float* __restrict__ p, float v, int n) {
    int i = blockIdx.x * blockDim.x + threadIdx.x;
    if (i < n) p[i] = v;
}

// ---------------------------------------------------------------------------
// Layer-1 projection: out[N,64] = X[N,K] @ W[K,64] + b   (K = 6 or 1, VALU)
// ---------------------------------------------------------------------------
__global__ void proj_small_kernel(const float* __restrict__ X, const float* __restrict__ W,
                                  const float* __restrict__ bias, float* __restrict__ out,
                                  int N, int K) {
    int idx = blockIdx.x * blockDim.x + threadIdx.x;
    if (idx >= N * 64) return;
    int n = idx >> 6, c = idx & 63;
    float acc = bias[c];
    const float* x = X + (size_t)n * K;
    for (int k = 0; k < K; ++k) acc += x[k] * W[k * 64 + c];
    out[idx] = acc;
}

// ---------------------------------------------------------------------------
// Layer-2 projection via WMMA: out[M,64] = A[M,64] @ W[64,64] + bias[64]
// 256-thread block = 8 waves, one 16-row strip per wave.
//  * W is loaded cooperatively ONCE per block, converted f32->f16 into a
//    transposed LDS copy WT[n][k] (row stride 72 halves: lanes hit 16
//    distinct 4-bank groups -> near conflict-free ds_load_b128 fragments).
//  * 4 column tiles x 2 K-steps of v_wmma_f32_16x16x32_f16, f32 accumulate.
//  * Epilogue staged through LDS (stride 68 floats), coalesced float4 stores.
//  * No early wave exit: __syncthreads is safe, EXEC all-1s at every WMMA.
// ---------------------------------------------------------------------------
#define WT_STRIDE 72
__global__ __launch_bounds__(256)
void gemm64_wmma_kernel(const float* __restrict__ A, const float* __restrict__ W,
                        const float* __restrict__ bias, float* __restrict__ out, int M) {
    __shared__ _Float16 WT[64 * WT_STRIDE];   // 9216 B
    __shared__ float    smem[8][16 * 68];     // 34816 B

    // ---- cooperative W transpose + f16 convert (one pass per block) ----
    for (int i = threadIdx.x; i < 64 * 64; i += 256) {   // i = k*64 + n
        int k = i >> 6, n = i & 63;
        WT[n * WT_STRIDE + k] = (_Float16)W[i];
    }
    __syncthreads();

    const int lane = threadIdx.x & 31;
    const int wib  = threadIdx.x >> 5;
    const int waveId = blockIdx.x * 8 + wib;
    int m0 = waveId * 16;
    const bool active = (m0 < M);
    if (!active) m0 = 0;                  // wave-uniform; tail waves do dummy work

    const int half = lane >> 4;           // 0: lanes 0-15, 1: lanes 16-31
    const int lr   = lane & 15;
    int row  = m0 + lr;
    int arow = (row < M) ? row : (M - 1); // per-lane clamp (cndmask, no EXEC change)

    // A fragments (16x32 f16): element i holds
    // k = half*16 + i -> contiguous 16-float run (matches B fragment order)
    v16h a0, a1;
    const float* Ar = A + (size_t)arow * 64;
#pragma unroll
    for (int i = 0; i < 16; ++i) {
        int k = half * 16 + i;
        a0[i] = (_Float16)Ar[k];
        a1[i] = (_Float16)Ar[k + 32];
    }

    float* st = smem[wib];
#pragma unroll
    for (int nt = 0; nt < 4; ++nt) {
        int ncol = nt * 16 + lr;
        // B fragments (32x16 f16): lanes 0-15 K=0..15, lanes 16-31 K=16..31,
        // N = lane%16 -> contiguous 16-half run in WT -> 2x ds_load_b128
        const _Float16* wt = &WT[ncol * WT_STRIDE];
        v16h b0, b1;
#pragma unroll
        for (int i = 0; i < 16; ++i) {
            b0[i] = wt[half * 16 + i];
            b1[i] = wt[32 + half * 16 + i];
        }
        v8f c = {};
        c = __builtin_amdgcn_wmma_f32_16x16x32_f16(false, a0, false, b0, (short)0, c, false, false);
        c = __builtin_amdgcn_wmma_f32_16x16x32_f16(false, a1, false, b1, (short)0, c, false, false);

        float bc = bias[ncol];            // C/D: column constant per lane
#pragma unroll
        for (int r = 0; r < 8; ++r)
            st[(half * 8 + r) * 68 + ncol] = c[r] + bc;   // conflict-free ds_store
    }
    __syncthreads();

    // Coalesced store: 16 rows x 16 float4 chunks, 8 chunks per lane
#pragma unroll
    for (int i = 0; i < 8; ++i) {
        int chunk = i * 32 + lane;
        int r = chunk >> 4, c4 = chunk & 15;
        int orow = m0 + r;
        if (active && orow < M) {
            float4 v = *(const float4*)&st[r * 68 + c4 * 4];
            *(float4*)&out[(size_t)orow * 64 + c4 * 4] = v;
        }
    }
}

// ---------------------------------------------------------------------------
// GATv2 edge passes (atomic 3-pass segment softmax)
// ---------------------------------------------------------------------------
__device__ __forceinline__ void atomicMaxF(float* addr, float val) {
    if (val >= 0.f) atomicMax((int*)addr, __float_as_int(val));
    else            atomicMin((unsigned int*)addr, __float_as_uint(val));
}

__device__ __forceinline__ void edge_sd(const int* src, const int* dst, int E, int t,
                                        int drop_self, int& s, int& d, bool& valid) {
    if (t < E) { s = src[t]; d = dst[t]; valid = !(drop_self && s == d); }
    else       { s = t - E;  d = t - E;  valid = true; }
}

// Pass 1: e = att . leaky_relu(hl[s] + hr[d]); atomicMax into emax[d]
__global__ void edge_score_max_kernel(const int* __restrict__ src, const int* __restrict__ dst,
                                      int E, int n_loop, int drop_self,
                                      const float* __restrict__ hl, const float* __restrict__ hr,
                                      const float* __restrict__ att, int H, int C,
                                      float* __restrict__ ebuf, float* __restrict__ emax) {
    int t = blockIdx.x * blockDim.x + threadIdx.x;
    if (t >= E + n_loop) return;
    int s, d; bool valid;
    edge_sd(src, dst, E, t, drop_self, s, d, valid);
    const float* pl = hl + (size_t)s * H * C;
    const float* pr = hr + (size_t)d * H * C;
    for (int h = 0; h < H; ++h) {
        float acc = 0.f;
        const float4* a4 = (const float4*)(att + h * C);
        const float4* l4 = (const float4*)(pl + h * C);
        const float4* r4 = (const float4*)(pr + h * C);
        for (int c = 0; c < C / 4; ++c) {
            float4 av = a4[c], lv = l4[c], rv = r4[c];
            float v0 = lv.x + rv.x, v1 = lv.y + rv.y, v2 = lv.z + rv.z, v3 = lv.w + rv.w;
            v0 = (v0 > 0.f) ? v0 : 0.2f * v0;
            v1 = (v1 > 0.f) ? v1 : 0.2f * v1;
            v2 = (v2 > 0.f) ? v2 : 0.2f * v2;
            v3 = (v3 > 0.f) ? v3 : 0.2f * v3;
            acc += av.x * v0 + av.y * v1 + av.z * v2 + av.w * v3;
        }
        ebuf[(size_t)t * H + h] = acc;
        if (valid) atomicMaxF(&emax[d * H + h], acc);
    }
}

// Pass 2: w = exp(e - emax[d]); atomicAdd into denom[d]; ebuf := w
__global__ void edge_exp_denom_kernel(const int* __restrict__ src, const int* __restrict__ dst,
                                      int E, int n_loop, int drop_self,
                                      const float* __restrict__ emax,
                                      float* __restrict__ ebuf, float* __restrict__ denom, int H) {
    int t = blockIdx.x * blockDim.x + threadIdx.x;
    if (t >= E + n_loop) return;
    int s, d; bool valid;
    edge_sd(src, dst, E, t, drop_self, s, d, valid);
    for (int h = 0; h < H; ++h) {
        float w = 0.f;
        if (valid) w = __expf(ebuf[(size_t)t * H + h] - emax[d * H + h]);
        ebuf[(size_t)t * H + h] = w;
        if (w != 0.f) atomicAdd(&denom[d * H + h], w);
    }
}

// Pass 3: out[d] += (w/denom[d]) * hl[s]  (one thread per edge x 4 channels)
__global__ void edge_aggr_kernel(const int* __restrict__ src, const int* __restrict__ dst,
                                 int E, int n_loop,
                                 const float* __restrict__ wbuf, const float* __restrict__ denom,
                                 const float* __restrict__ hl, float* __restrict__ outacc,
                                 int H, int C) {
    int idx = blockIdx.x * blockDim.x + threadIdx.x;
    int HC4 = (H * C) >> 2;
    int total = (E + n_loop) * HC4;
    if (idx >= total) return;
    int t = idx / HC4, q = idx - t * HC4;      // q: float4 chunk within HC row
    int h = (q << 2) / C;                      // chunk never straddles heads (C%4==0)
    float w = wbuf[(size_t)t * H + h];
    if (w == 0.f) return;
    int s, d;
    if (t < E) { s = src[t]; d = dst[t]; } else { s = t - E; d = t - E; }
    float alpha = w / (denom[d * H + h] + 1e-16f);
    float4 v = ((const float4*)(hl + (size_t)s * H * C))[q];
    float* o = outacc + (size_t)d * H * C + (q << 2);
    atomicAdd(o + 0, alpha * v.x);
    atomicAdd(o + 1, alpha * v.y);
    atomicAdd(o + 2, alpha * v.z);
    atomicAdd(o + 3, alpha * v.w);
}

// ---------------------------------------------------------------------------
// Epilogues: elu(A + ba [+ B + bb])  and  LayerNorm(A + ba [+ B + bb])
// ---------------------------------------------------------------------------
__global__ void combine_elu_kernel(const float* __restrict__ A, const float* __restrict__ ba,
                                   const float* __restrict__ B, const float* __restrict__ bb,
                                   float* __restrict__ out, int N) {
    int idx = blockIdx.x * blockDim.x + threadIdx.x;      // over N*16 float4s
    if (idx >= N * 16) return;
    int j4 = idx & 15;
    float4 x = ((const float4*)A)[idx];
    float4 c = ((const float4*)ba)[j4];
    x.x += c.x; x.y += c.y; x.z += c.z; x.w += c.w;
    if (B) {
        float4 y = ((const float4*)B)[idx];
        float4 e = ((const float4*)bb)[j4];
        x.x += y.x + e.x; x.y += y.y + e.y; x.z += y.z + e.z; x.w += y.w + e.w;
    }
    x.x = (x.x > 0.f) ? x.x : (__expf(x.x) - 1.f);
    x.y = (x.y > 0.f) ? x.y : (__expf(x.y) - 1.f);
    x.z = (x.z > 0.f) ? x.z : (__expf(x.z) - 1.f);
    x.w = (x.w > 0.f) ? x.w : (__expf(x.w) - 1.f);
    ((float4*)out)[idx] = x;
}

__global__ void combine_ln_kernel(const float* __restrict__ A, const float* __restrict__ ba,
                                  const float* __restrict__ B, const float* __restrict__ bb,
                                  const float* __restrict__ w, const float* __restrict__ b,
                                  float* __restrict__ out, int N) {
    int r = blockIdx.x * blockDim.x + threadIdx.x;
    if (r >= N) return;
    const float4* Ar = (const float4*)(A + (size_t)r * 64);
    const float4* Br = B ? (const float4*)(B + (size_t)r * 64) : nullptr;
    const float4* ba4 = (const float4*)ba;
    const float4* bb4 = (const float4*)bb;
    float x[64];
    float mu = 0.f;
#pragma unroll
    for (int j = 0; j < 16; ++j) {
        float4 v = Ar[j], c = ba4[j];
        float4 t = make_float4(v.x + c.x, v.y + c.y, v.z + c.z, v.w + c.w);
        if (Br) { float4 y = Br[j], e = bb4[j];
                  t.x += y.x + e.x; t.y += y.y + e.y; t.z += y.z + e.z; t.w += y.w + e.w; }
        x[4*j+0] = t.x; x[4*j+1] = t.y; x[4*j+2] = t.z; x[4*j+3] = t.w;
        mu += t.x + t.y + t.z + t.w;
    }
    mu *= (1.f / 64.f);
    float var = 0.f;
#pragma unroll
    for (int j = 0; j < 64; ++j) { float t = x[j] - mu; var += t * t; }
    var *= (1.f / 64.f);
    float sc = rsqrtf(var + 1e-5f);
    float* o = out + (size_t)r * 64;
#pragma unroll
    for (int j = 0; j < 64; ++j) o[j] = (x[j] - mu) * sc * w[j] + b[j];
}

// ---------------------------------------------------------------------------
// Host-side conv driver (all on `stream`; safe for graph capture)
// ---------------------------------------------------------------------------
static void run_gatv2(hipStream_t stream, const int* src, const int* dst, int E, int n_loop,
                      int drop_self, const float* hl, const float* hr, const float* att,
                      int H, int C, int n_dst,
                      float* ebuf, float* emax, float* denom, float* outacc) {
    int Et = E + n_loop;
    fill_kernel<<<CDIV(n_dst * H, 256), 256, 0, stream>>>(emax, -1e30f, n_dst * H);
    fill_kernel<<<CDIV(n_dst * H, 256), 256, 0, stream>>>(denom, 0.f, n_dst * H);
    fill_kernel<<<CDIV(n_dst * H * C, 256), 256, 0, stream>>>(outacc, 0.f, n_dst * H * C);
    edge_score_max_kernel<<<CDIV(Et, 256), 256, 0, stream>>>(src, dst, E, n_loop, drop_self,
                                                             hl, hr, att, H, C, ebuf, emax);
    edge_exp_denom_kernel<<<CDIV(Et, 256), 256, 0, stream>>>(src, dst, E, n_loop, drop_self,
                                                             emax, ebuf, denom, H);
    int tot = Et * ((H * C) >> 2);
    edge_aggr_kernel<<<CDIV(tot, 256), 256, 0, stream>>>(src, dst, E, n_loop, ebuf, denom,
                                                         hl, outacc, H, C);
}

extern "C" void kernel_launch(void* const* d_in, const int* in_sizes, int n_in,
                              void* d_out, int out_size, void* d_ws, size_t ws_size,
                              hipStream_t stream) {
    const float* x_veh = (const float*)d_in[0];
    const float* x_rsu = (const float*)d_in[1];
    const int* ei_vv = (const int*)d_in[2];
    const int* ei_vi = (const int*)d_in[3];
    const int* ei_iv = (const int*)d_in[4];
    const int NV  = in_sizes[0] / 6;
    const int NR  = in_sizes[1];
    const int EVV = in_sizes[2] / 2;
    const int EVI = in_sizes[3] / 2;
    const int EIV = in_sizes[4] / 2;

    const float* P[45];
    for (int i = 5; i < 45; ++i) P[i] = (const float*)d_in[i];
    const float *p1v_Wl = P[5],  *p1v_bl = P[6],  *p1v_Wr = P[7],  *p1v_br = P[8],  *p1v_att = P[9],  *p1v_b = P[10];
    const float *p1i_Wl = P[11], *p1i_bl = P[12], *p1i_Wr = P[13], *p1i_br = P[14], *p1i_att = P[15], *p1i_b = P[16];
    const float *p1r_Wl = P[17], *p1r_bl = P[18], *p1r_Wr = P[19], *p1r_br = P[20], *p1r_att = P[21], *p1r_b = P[22];
    const float *p2v_Wl = P[23], *p2v_bl = P[24], *p2v_Wr = P[25], *p2v_br = P[26], *p2v_att = P[27], *p2v_b = P[28];
    const float *p2i_Wl = P[29], *p2i_bl = P[30], *p2i_Wr = P[31], *p2i_br = P[32], *p2i_att = P[33], *p2i_b = P[34];
    const float *p2r_Wl = P[35], *p2r_bl = P[36], *p2r_Wr = P[37], *p2r_br = P[38], *p2r_att = P[39], *p2r_b = P[40];
    const float *lnv_w = P[41], *lnv_b = P[42], *lnr_w = P[43], *lnr_b = P[44];

    float* out = (float*)d_out;

    // ---- workspace layout (floats; all offsets multiples of 64 for float4) ----
    size_t big = (size_t)NV * 64, sml = (size_t)NR * 64;
    float* ws   = (float*)d_ws;
    float* T0   = ws;             float* T1 = T0 + big;  float* T2 = T1 + big;  float* T3 = T2 + big;
    float* S0   = T3 + big;       float* S1 = S0 + sml;
    float* Hveh = S1 + sml;       float* Hrsu = Hveh + big;
    float* A0   = Hrsu + sml;     float* A1 = A0 + big;  float* Asm = A1 + big;
    float* EW   = Asm + sml;                       // edge e/w buffer: max (EVV+NV)*2
    float* EMv  = EW + (size_t)(EVV + NV) * 2;
    float* DNv  = EMv + (size_t)NV * 2;
    float* EMr  = DNv + (size_t)NV * 2;
    float* DNr  = EMr + (size_t)NR * 2;
    (void)ws_size; (void)n_in; (void)out_size;

    // ================= Layer 1 (heads=2, hid=32) =================
    proj_small_kernel<<<CDIV(NV * 64, 256), 256, 0, stream>>>(x_veh, p1v_Wl, p1v_bl, T0, NV, 6);
    proj_small_kernel<<<CDIV(NV * 64, 256), 256, 0, stream>>>(x_veh, p1v_Wr, p1v_br, T1, NV, 6);
    proj_small_kernel<<<CDIV(NV * 64, 256), 256, 0, stream>>>(x_veh, p1i_Wl, p1i_bl, T2, NV, 6);
    proj_small_kernel<<<CDIV(NR * 64, 256), 256, 0, stream>>>(x_rsu, p1i_Wr, p1i_br, S0, NR, 1);
    proj_small_kernel<<<CDIV(NR * 64, 256), 256, 0, stream>>>(x_rsu, p1r_Wl, p1r_bl, S1, NR, 1);
    proj_small_kernel<<<CDIV(NV * 64, 256), 256, 0, stream>>>(x_veh, p1r_Wr, p1r_br, T3, NV, 6);

    run_gatv2(stream, ei_vv, ei_vv + EVV, EVV, NV, 1, T0, T1, p1v_att, 2, 32, NV, EW, EMv, DNv, A0);
    run_gatv2(stream, ei_vi, ei_vi + EVI, EVI, 0,  0, T2, S0, p1i_att, 2, 32, NR, EW, EMr, DNr, Asm);
    run_gatv2(stream, ei_iv, ei_iv + EIV, EIV, 0,  0, S1, T3, p1r_att, 2, 32, NV, EW, EMv, DNv, A1);

    combine_elu_kernel<<<CDIV(NV * 16, 256), 256, 0, stream>>>(A0, p1v_b, A1, p1r_b, Hveh, NV);
    combine_elu_kernel<<<CDIV(NR * 16, 256), 256, 0, stream>>>(Asm, p1i_b, nullptr, nullptr, Hrsu, NR);

    // ================= Layer 2 (heads=1, out=64) — WMMA projections =========
    {
        int gv = CDIV(CDIV(NV, 16), 8), gr = CDIV(CDIV(NR, 16), 8);
        gemm64_wmma_kernel<<<gv, 256, 0, stream>>>(Hveh, p2v_Wl, p2v_bl, T0, NV);
        gemm64_wmma_kernel<<<gv, 256, 0, stream>>>(Hveh, p2v_Wr, p2v_br, T1, NV);
        gemm64_wmma_kernel<<<gv, 256, 0, stream>>>(Hveh, p2i_Wl, p2i_bl, T2, NV);
        gemm64_wmma_kernel<<<gr, 256, 0, stream>>>(Hrsu, p2i_Wr, p2i_br, S0, NR);
        gemm64_wmma_kernel<<<gr, 256, 0, stream>>>(Hrsu, p2r_Wl, p2r_bl, S1, NR);
        gemm64_wmma_kernel<<<gv, 256, 0, stream>>>(Hveh, p2r_Wr, p2r_br, T3, NV);
    }

    run_gatv2(stream, ei_vv, ei_vv + EVV, EVV, NV, 1, T0, T1, p2v_att, 1, 64, NV, EW, EMv, DNv, A0);
    run_gatv2(stream, ei_iv, ei_iv + EIV, EIV, NR, 1, S1, T3, p2r_att, 1, 64, NV, EW, EMv, DNv, A1);
    run_gatv2(stream, ei_vi, ei_vi + EVI, EVI, NR, 1, T2, S0, p2i_att, 1, 64, NR, EW, EMr, DNr, Asm);

    combine_ln_kernel<<<CDIV(NV, 128), 128, 0, stream>>>(A0, p2v_b, A1, p2r_b, lnv_w, lnv_b, out, NV);
    combine_ln_kernel<<<CDIV(NR, 128), 128, 0, stream>>>(Asm, p2i_b, nullptr, nullptr, lnr_w, lnr_b,
                                                         out + (size_t)NV * 64, NR);
}
